// BackboneRITS_51058571215206
// MI455X (gfx1250) — matrix operand order repaired and checked by reading, not compiled, not run on hardware
//
#include <hip/hip_runtime.h>
#include <hip/hip_bf16.h>

#define B_  1024
#define T_  128
#define F_  128
#define H_  512
#define G4  (4*H_)   // 2048
#define F2  (2*F_)   // 256

typedef __attribute__((ext_vector_type(16))) __bf16 v16bf;
typedef __attribute__((ext_vector_type(8)))  float  v8f;
typedef __attribute__((ext_vector_type(4)))  unsigned int v4u;

// bf16 weight workspace layout (element offsets, 2 bytes each)
#define OFF_WGH   0                     // [H , F ] 512x128
#define OFF_WHIST (OFF_WGH   + H_*F_)   // [F , H ] 128x512
#define OFF_WFEAT (OFF_WHIST + F_*H_)   // [F , F ] (diag zeroed)
#define OFF_WCOMB (OFF_WFEAT + F_*F_)   // [F , 2F]
#define OFF_WIH   (OFF_WCOMB + F_*F2)   // [4H, 2F]
#define OFF_WHH   (OFF_WIH   + G4*F2)   // [4H, H ]
#define WS_ELEMS  (OFF_WHH   + G4*H_)   // 1,753,088 bf16 elems (byte offset 16B-aligned)

__device__ __forceinline__ unsigned short f2bf(float f) {
  unsigned int u = __float_as_uint(f);
  unsigned int r = u + 0x7FFFu + ((u >> 16) & 1u);   // round-to-nearest-even
  return (unsigned short)(r >> 16);
}

__device__ __forceinline__ float sigm(float x) { return 1.0f / (1.0f + __expf(-x)); }

union FragBF { v16bf v; v4u q[2]; };

// A-matrix fragment (16xK, 16-bit, row-major src with leading dim ld).
// lane<16: row=lane, K = kb+{0..7} then kb+{16..23}; lane>=16: row=lane-16, K = kb+{8..15} then kb+{24..31}
__device__ __forceinline__ v16bf frag_a(const unsigned short* p, int ld, int kb, int lane) {
  int row = lane & 15;
  int kf  = kb + ((lane & 16) ? 8 : 0);
  const unsigned short* q = p + row * ld + kf;
  FragBF f;
  f.q[0] = *(const v4u*)(q);
  f.q[1] = *(const v4u*)(q + 16);
  return f.v;
}

// B-matrix fragment (KxN tile) taken from row-major W[N,K]: column n of B = row n of W.
// lane<16: N=lane, K = kb+{0..7} then kb+{16..23}; lane>=16: N=lane-16, K = kb+{8..15} then kb+{24..31}
__device__ __forceinline__ v16bf frag_b(const unsigned short* p, int ld, int nb, int kb, int lane) {
  int n  = nb + (lane & 15);
  int kf = kb + ((lane & 16) ? 8 : 0);
  const unsigned short* q = p + n * ld + kf;
  FragBF f;
  f.q[0] = *(const v4u*)(q);
  f.q[1] = *(const v4u*)(q + 16);
  return f.v;
}

__device__ __forceinline__ v8f wmma_bf16(v16bf a, v16bf b, v8f c) {
  return __builtin_amdgcn_wmma_f32_16x16x32_bf16(false, a, false, b, (short)0, c, false, false);
}

// ---------------- weight conversion to bf16 (with masking) + fused bias ----------------
__global__ void rits_prep(const float* __restrict__ Wgh, const float* __restrict__ Whist,
                          const float* __restrict__ Wfeat, const float* __restrict__ Wcomb,
                          const float* __restrict__ Wih, const float* __restrict__ Whh,
                          const float* __restrict__ b_ih, const float* __restrict__ b_hh,
                          unsigned short* __restrict__ wbf) {
  int i = blockIdx.x * blockDim.x + threadIdx.x;
  int stride = gridDim.x * blockDim.x;
  float* bsum = (float*)(wbf + WS_ELEMS);
  for (int idx = i; idx < WS_ELEMS + G4; idx += stride) {
    if (idx >= WS_ELEMS) {                       // fused LSTM bias, fp32
      int j = idx - WS_ELEMS;
      bsum[j] = b_ih[j] + b_hh[j];
      continue;
    }
    float v;
    if (idx < OFF_WHIST)      v = Wgh[idx - OFF_WGH];
    else if (idx < OFF_WFEAT) v = Whist[idx - OFF_WHIST];
    else if (idx < OFF_WCOMB) {
      int j = idx - OFF_WFEAT; int r = j >> 7, c = j & 127;
      v = (r == c) ? 0.0f : Wfeat[j];            // FeatureRegression: zero diagonal
    }
    else if (idx < OFF_WIH)   v = Wcomb[idx - OFF_WCOMB];
    else if (idx < OFF_WHH)   v = Wih[idx - OFF_WIH];
    else                      v = Whh[idx - OFF_WHH];
    wbf[idx] = f2bf(v);
  }
}

// ---------------- persistent recurrent kernel: 16 batch rows / workgroup ----------------
// Allow up to 512 VGPRs/wave (>=2 waves/SIMD): lets the compiler keep phase A-D weight
// fragments register-resident across the t-loop (round-1 regime, no scratch spills),
// while guaranteeing occupancy never drops below 2 waves/SIMD.
__global__ __launch_bounds__(256, 2)
void rits_main(const float* __restrict__ X, const float* __restrict__ M,
               const float* __restrict__ D,
               const float* __restrict__ b_gh, const float* __restrict__ W_gx,
               const float* __restrict__ b_gx, const float* __restrict__ b_hist,
               const float* __restrict__ b_feat, const float* __restrict__ b_comb,
               const unsigned short* __restrict__ wbf,
               float* __restrict__ o_imp, float* __restrict__ o_ch1,
               float* __restrict__ o_hT, float* __restrict__ o_xh,
               float* __restrict__ o_ch2, float* __restrict__ o_zh)
{
  __shared__ __align__(16) float          sh_h [16 * H_];   // fp32 h state
  __shared__ __align__(16) float          sh_c [16 * H_];   // fp32 c state
  __shared__ __align__(16) unsigned short sh_hbf[16 * H_];  // decayed h, bf16 (GEMM A)
  __shared__ __align__(16) float          sh_x [16 * F_];
  __shared__ __align__(16) float          sh_m [16 * F_];
  __shared__ __align__(16) unsigned short sh_dbf[16 * F_];
  __shared__ __align__(16) float          sh_xh[16 * F_];
  __shared__ __align__(16) unsigned short sh_xcbf[16 * F_];
  __shared__ __align__(16) unsigned short sh_comb[16 * F2]; // [gamma_x | m] bf16
  __shared__ __align__(16) unsigned short sh_gin [16 * F2]; // [c_c | m] bf16

  const int tid  = threadIdx.x;
  const int lane = tid & 31;
  const int wave = tid >> 5;
  const int b0   = blockIdx.x * 16;

  const unsigned short* Wgh   = wbf + OFF_WGH;
  const unsigned short* Whist = wbf + OFF_WHIST;
  const unsigned short* Wfeat = wbf + OFF_WFEAT;
  const unsigned short* Wcomb = wbf + OFF_WCOMB;
  const unsigned short* Wih   = wbf + OFF_WIH;
  const unsigned short* Whh   = wbf + OFF_WHH;
  const float*          bsum  = (const float*)(wbf + WS_ELEMS);

  for (int i = tid; i < 16 * H_; i += 256) { sh_h[i] = 0.0f; sh_c[i] = 0.0f; }

  // hoist the gamma_x diagonal params for this thread's 8 fixed (row,f) elements
  float wdiag[8], bgx[8];
  #pragma unroll
  for (int i = 0; i < 8; ++i) {
    int f = (tid + i * 256) & 127;
    wdiag[i] = W_gx[f * F_ + f];
    bgx[i]   = b_gx[f];
  }
  __syncthreads();

  for (int t = 0; t < T_; ++t) {
    // ---------- load x,m,d; elementwise gamma_x ----------
    #pragma unroll
    for (int i = 0; i < 8; ++i) {
      int idx = tid + i * 256;
      int r = idx >> 7, f = idx & 127;
      long long g = ((long long)(b0 + r) * T_ + t) * F_ + f;
      float x = X[g], mm = M[g], d = D[g];
      sh_x[idx] = x;
      sh_m[idx] = mm;
      sh_dbf[idx] = f2bf(d);
      float gx = __expf(-fmaxf(fmaf(d, wdiag[i], bgx[i]), 0.0f));
      unsigned short mb = f2bf(mm);
      sh_comb[r * F2 + f]      = f2bf(gx);
      sh_comb[r * F2 + F_ + f] = mb;
      sh_gin [r * F2 + F_ + f] = mb;
    }
    __syncthreads();

    // ---------- phase A: gamma_h = exp(-relu(d@Wghᵀ+b)); h_bf = bf16(h*gamma_h) ----------
    for (int i = 0; i < 4; ++i) {
      int n0 = (wave * 4 + i) * 16;
      v8f acc = {0.f,0.f,0.f,0.f,0.f,0.f,0.f,0.f};
      #pragma unroll
      for (int kb = 0; kb < F_; kb += 32)
        acc = wmma_bf16(frag_a(sh_dbf, F_, kb, lane), frag_b(Wgh, F_, n0, kb, lane), acc);
      int n = n0 + (lane & 15), mbs = (lane >> 4) * 8;
      float bias = b_gh[n];
      #pragma unroll
      for (int v = 0; v < 8; ++v) {
        int m = mbs + v;
        float g = __expf(-fmaxf(acc[v] + bias, 0.0f));
        sh_hbf[m * H_ + n] = f2bf(sh_h[m * H_ + n] * g);
      }
    }
    __syncthreads();

    // ---------- phase B: x_h = h@W_histᵀ + b; x_c = m*x + (1-m)*x_h ----------
    {
      int n0 = wave * 16;
      v8f acc = {0.f,0.f,0.f,0.f,0.f,0.f,0.f,0.f};
      #pragma unroll 4
      for (int kb = 0; kb < H_; kb += 32)
        acc = wmma_bf16(frag_a(sh_hbf, H_, kb, lane), frag_b(Whist, H_, n0, kb, lane), acc);
      int n = n0 + (lane & 15), mbs = (lane >> 4) * 8;
      float bias = b_hist[n];
      #pragma unroll
      for (int v = 0; v < 8; ++v) {
        int m = mbs + v;
        float xh = acc[v] + bias;
        sh_xh[m * F_ + n] = xh;
        o_xh[((long long)(b0 + m) * T_ + t) * F_ + n] = xh;
        float mm = sh_m[m * F_ + n];
        sh_xcbf[m * F_ + n] = f2bf(mm * sh_x[m * F_ + n] + (1.0f - mm) * xh);
      }
    }
    __syncthreads();

    // ---------- phase C+D: z_h, alpha, c_h, c_c; write ZH/CH/imputed ----------
    {
      int n0 = wave * 16;
      v8f accz = {0.f,0.f,0.f,0.f,0.f,0.f,0.f,0.f};
      #pragma unroll
      for (int kb = 0; kb < F_; kb += 32)
        accz = wmma_bf16(frag_a(sh_xcbf, F_, kb, lane), frag_b(Wfeat, F_, n0, kb, lane), accz);
      v8f acca = {0.f,0.f,0.f,0.f,0.f,0.f,0.f,0.f};
      #pragma unroll
      for (int kb = 0; kb < F2; kb += 32)
        acca = wmma_bf16(frag_a(sh_comb, F2, kb, lane), frag_b(Wcomb, F2, n0, kb, lane), acca);
      int n = n0 + (lane & 15), mbs = (lane >> 4) * 8;
      float bz = b_feat[n], bc = b_comb[n];
      #pragma unroll
      for (int v = 0; v < 8; ++v) {
        int m = mbs + v;
        float z  = accz[v] + bz;
        float al = sigm(acca[v] + bc);
        float xh = sh_xh[m * F_ + n];
        float ch = al * z + (1.0f - al) * xh;
        float mm = sh_m[m * F_ + n];
        float cc = mm * sh_x[m * F_ + n] + (1.0f - mm) * ch;
        long long g = ((long long)(b0 + m) * T_ + t) * F_ + n;
        o_zh[g] = z; o_ch1[g] = ch; o_ch2[g] = ch; o_imp[g] = cc;
        sh_gin[m * F2 + n] = f2bf(cc);
      }
    }
    __syncthreads();

    // ---------- phase E: gates = [c_c,m]@W_ihᵀ + h@W_hhᵀ + b; LSTM update ----------
    for (int ct = 0; ct < 4; ++ct) {
      int n0 = wave * 64 + ct * 16;
      v8f ai = {0.f,0.f,0.f,0.f,0.f,0.f,0.f,0.f};
      v8f af = ai, ag = ai, ao = ai;
      #pragma unroll 2
      for (int kb = 0; kb < F2; kb += 32) {
        v16bf a = frag_a(sh_gin, F2, kb, lane);
        ai = wmma_bf16(a, frag_b(Wih, F2, n0,         kb, lane), ai);
        af = wmma_bf16(a, frag_b(Wih, F2, n0 + H_,    kb, lane), af);
        ag = wmma_bf16(a, frag_b(Wih, F2, n0 + 2*H_,  kb, lane), ag);
        ao = wmma_bf16(a, frag_b(Wih, F2, n0 + 3*H_,  kb, lane), ao);
      }
      #pragma unroll 2
      for (int kb = 0; kb < H_; kb += 32) {
        v16bf a = frag_a(sh_hbf, H_, kb, lane);
        ai = wmma_bf16(a, frag_b(Whh, H_, n0,         kb, lane), ai);
        af = wmma_bf16(a, frag_b(Whh, H_, n0 + H_,    kb, lane), af);
        ag = wmma_bf16(a, frag_b(Whh, H_, n0 + 2*H_,  kb, lane), ag);
        ao = wmma_bf16(a, frag_b(Whh, H_, n0 + 3*H_,  kb, lane), ao);
      }
      int n = n0 + (lane & 15), mbs = (lane >> 4) * 8;
      float bi = bsum[n];
      float bf = bsum[n + H_];
      float bg = bsum[n + 2*H_];
      float bo = bsum[n + 3*H_];
      #pragma unroll
      for (int v = 0; v < 8; ++v) {
        int m = mbs + v;
        float ig = sigm(ai[v] + bi);
        float fg = sigm(af[v] + bf);
        float gg = tanhf(ag[v] + bg);
        float og = sigm(ao[v] + bo);
        float cnew = fg * sh_c[m * H_ + n] + ig * gg;
        float hnew = og * tanhf(cnew);
        sh_c[m * H_ + n] = cnew;
        sh_h[m * H_ + n] = hnew;
        if (t == T_ - 1) o_hT[(long long)(b0 + m) * H_ + n] = hnew;
      }
    }
    __syncthreads();
  }
}

extern "C" void kernel_launch(void* const* d_in, const int* in_sizes, int n_in,
                              void* d_out, int out_size, void* d_ws, size_t ws_size,
                              hipStream_t stream) {
  const float* X      = (const float*)d_in[0];
  const float* Mm     = (const float*)d_in[1];
  const float* D      = (const float*)d_in[2];
  const float* W_gh   = (const float*)d_in[3];
  const float* b_gh   = (const float*)d_in[4];
  const float* W_gx   = (const float*)d_in[5];
  const float* b_gx   = (const float*)d_in[6];
  const float* W_hist = (const float*)d_in[7];
  const float* b_hist = (const float*)d_in[8];
  const float* W_feat = (const float*)d_in[9];
  const float* b_feat = (const float*)d_in[10];
  const float* W_comb = (const float*)d_in[11];
  const float* b_comb = (const float*)d_in[12];
  const float* W_ih   = (const float*)d_in[13];
  const float* W_hh   = (const float*)d_in[14];
  const float* b_ih   = (const float*)d_in[15];
  const float* b_hh   = (const float*)d_in[16];

  unsigned short* wbf = (unsigned short*)d_ws;
  float* out = (float*)d_out;
  const long long BTF = (long long)B_ * T_ * F_;
  float* o_imp = out;                             // imputed
  float* o_ch1 = out + BTF;                       // reconstruction (CH)
  float* o_hT  = out + 2 * BTF;                   // h_T
  float* o_xh  = o_hT + (long long)B_ * H_;       // XH
  float* o_ch2 = o_xh + BTF;                      // CH (again)
  float* o_zh  = o_ch2 + BTF;                     // ZH

  rits_prep<<<512, 256, 0, stream>>>(W_gh, W_hist, W_feat, W_comb, W_ih, W_hh,
                                     b_ih, b_hh, wbf);
  rits_main<<<B_ / 16, 256, 0, stream>>>(X, Mm, D, b_gh, W_gx, b_gx, b_hist, b_feat,
                                         b_comb, wbf,
                                         o_imp, o_ch1, o_hT, o_xh, o_ch2, o_zh);
}
